// PhysicsGAT_38568806318222
// MI455X (gfx1250) — matrix operand order repaired and checked by reading, not compiled, not run on hardware
//
#include <hip/hip_runtime.h>
#include <hip/hip_bf16.h>
#include <math.h>

#define IN_F   25
#define HID_F  128
#define OUT_F  64
#define HEADS  4
#define ED_F   6

typedef __attribute__((ext_vector_type(16))) _Float16 v16h;
typedef __attribute__((ext_vector_type(8)))  _Float16 v8h;
typedef __attribute__((ext_vector_type(8)))  float    v8f;

// ---------- order-preserving float<->uint mapping for atomic max ----------
__device__ __forceinline__ unsigned mapf(float f) {
  unsigned u = __float_as_uint(f);
  return (u & 0x80000000u) ? ~u : (u | 0x80000000u);
}
__device__ __forceinline__ float unmapf(unsigned u) {
  return (u & 0x80000000u) ? __uint_as_float(u ^ 0x80000000u)
                           : __uint_as_float(~u);
}

// ---------- WMMA GEMM ----------
// Out[Npad,P] = Ah[Npad,KPAD] (f16, zero-padded) x Wt[P,KPAD] (f16 = W^T, zero-padded).
// One wave computes a 64x16 strip (4 M-tiles sharing each B tile). KPAD/P are
// compile-time so the K loop fully unrolls and all addressing is 32-bit with
// immediate store offsets.
template <int KPAD, int P>
__global__ __launch_bounds__(128) void k_gemm_wmma(const _Float16* __restrict__ Ah,
                                                   const _Float16* __restrict__ Wt,
                                                   float* __restrict__ Out,
                                                   int Npad) {
  const int lane = threadIdx.x & 31;
  const int wave = threadIdx.x >> 5;
  const int wpb  = blockDim.x >> 5;
  const int strips = Npad >> 6;                       // 64-row strips
  const int tilesN = P / 16;
  const int sid = blockIdx.x * wpb + wave;
  if (sid >= strips * tilesN) return;                 // wave-uniform
  const int sm = sid / tilesN;
  const int tn = sid % tilesN;
  const bool hi = lane >= 16;
  const int l16 = lane & 15;
  const int col = tn * 16 + l16;

  // B: lane = column `col`; elements 0..15 <-> K = k0 + (hi?16:0) + e  (contiguous)
  const _Float16* bp = Wt + (unsigned)(col * KPAD + (hi ? 16 : 0));
  // A: lane = row (within tile); elems 0..15 <-> K = k0+(hi?16:0)+e  (contiguous)
  const _Float16* ap = Ah + (unsigned)((sm * 64 + l16) * KPAD + (hi ? 16 : 0));

  v8f c0 = {}, c1 = {}, c2 = {}, c3 = {};
#pragma unroll
  for (int k0 = 0; k0 < KPAD; k0 += 32) {
    v16h b = *(const v16h*)(bp + k0);                 // 32B contiguous, 32B aligned
#pragma unroll
    for (int mi = 0; mi < 4; ++mi) {
      const _Float16* ar = ap + mi * 16 * KPAD;
      v8h alo = *(const v8h*)(ar + k0);
      v8h ahi = *(const v8h*)(ar + k0 + 8);
      v16h a = __builtin_shufflevector(alo, ahi, 0, 1, 2, 3, 4, 5, 6, 7,
                                       8, 9, 10, 11, 12, 13, 14, 15);
      v8f* cc = (mi == 0) ? &c0 : (mi == 1) ? &c1 : (mi == 2) ? &c2 : &c3;
      *cc = __builtin_amdgcn_wmma_f32_16x16x32_f16(false, a, false, b,
                                                   (short)0, *cc, false, false);
    }
  }
#pragma unroll
  for (int mi = 0; mi < 4; ++mi) {
    const v8f c = (mi == 0) ? c0 : (mi == 1) ? c1 : (mi == 2) ? c2 : c3;
    // single 32-bit element offset; per-v offsets are compile-time (imm offsets)
    float* op = Out + (unsigned)((sm * 64 + mi * 16 + (hi ? 8 : 0)) * P + col);
#pragma unroll
    for (int v = 0; v < 8; ++v)
      op[v * P] = c[v];
  }
}

// ---------- f32 -> padded f16 conversion (layer-1 node features) ----------
__global__ void k_cvt_x(const float* __restrict__ x, _Float16* __restrict__ Ah,
                        int N, int Npad) {
  int idx = blockIdx.x * blockDim.x + threadIdx.x;      // over Npad*32
  if (idx >= Npad * 32) return;
  int n = idx >> 5, k = idx & 31;
  float v = (n < N && k < IN_F) ? x[(size_t)n * IN_F + k] : 0.0f;
  Ah[idx] = (_Float16)v;
}

// ---------- W[K,P] f32 -> Wt[P,KPAD] f16 (transpose + zero-pad) ----------
__global__ void k_cvt_w(const float* __restrict__ W, _Float16* __restrict__ Wt,
                        int K, int P, int KPAD) {
  int idx = blockIdx.x * blockDim.x + threadIdx.x;      // over P*KPAD
  if (idx >= P * KPAD) return;
  int p = idx / KPAD, k = idx % KPAD;
  Wt[idx] = (_Float16)((k < K) ? W[(size_t)k * P + p] : 0.0f);
}

// ---------- edge_attr column sums ----------
__global__ void k_ea_sum(const float* __restrict__ ea, float* __restrict__ eaSum, int E) {
  float acc[ED_F] = {};
  for (int i = blockIdx.x * blockDim.x + threadIdx.x; i < E;
       i += gridDim.x * blockDim.x)
#pragma unroll
    for (int k = 0; k < ED_F; ++k) acc[k] += ea[(size_t)i * ED_F + k];
#pragma unroll
  for (int k = 0; k < ED_F; ++k) atomicAdd(&eaSum[k], acc[k]);
}

// ---------- fold We/att_e into w_eff[6,H] and self-loop attention a_loop[H] ----------
__global__ void k_weff(const float* __restrict__ We, const float* __restrict__ attE,
                       const float* __restrict__ eaSum, float* __restrict__ wEff,
                       float* __restrict__ aLoop, int Hh, int C, int E) {
  int t = threadIdx.x;
  if (t < ED_F * Hh) {
    int d = t / Hh, h = t % Hh;
    float s = 0.f;
    for (int c = 0; c < C; ++c) s += We[(size_t)d * (Hh * C) + h * C + c] * attE[h * C + c];
    wEff[d * Hh + h] = s;
  }
  __syncthreads();
  if (t < Hh) {
    float s = 0.f;
    for (int d = 0; d < ED_F; ++d) s += (eaSum[d] / (float)E) * wEff[d * Hh + t];
    aLoop[t] = s;
  }
}

// ---------- per-node attention scalars ----------
__global__ void k_node_attn(const float* __restrict__ hfeat, const float* __restrict__ attS,
                            const float* __restrict__ attD, float* __restrict__ aS,
                            float* __restrict__ aD, int N, int Hh, int C) {
  int idx = blockIdx.x * blockDim.x + threadIdx.x;
  if (idx >= N * Hh) return;
  int h = idx % Hh, n = idx / Hh;
  const float* hv = hfeat + (size_t)n * (Hh * C) + h * C;
  float ss = 0.f, sd = 0.f;
  for (int c = 0; c < C; ++c) { float v = hv[c]; ss += v * attS[h * C + c]; sd += v * attD[h * C + c]; }
  aS[idx] = ss; aD[idx] = sd;
}

__global__ void k_init_bias(float* __restrict__ out, const float* __restrict__ bias,
                            int N, int P) {
  size_t idx = (size_t)blockIdx.x * blockDim.x + threadIdx.x;
  if (idx >= (size_t)N * P) return;
  out[idx] = bias[idx % P];
}

__device__ __forceinline__ float edge_alpha(int h, int Hh, int s, int d,
                                            const float* eav, bool isLoop,
                                            const float* __restrict__ aS,
                                            const float* __restrict__ aD,
                                            const float* __restrict__ wEff,
                                            const float* __restrict__ aLoop) {
  float ae;
  if (isLoop) ae = aLoop[h];
  else {
    ae = 0.f;
#pragma unroll
    for (int k = 0; k < ED_F; ++k) ae += eav[k] * wEff[k * Hh + h];
  }
  float al = aS[s * Hh + h] + aD[d * Hh + h] + ae;
  return al > 0.f ? al : 0.2f * al;           // leaky_relu 0.2
}

// ---------- pass 1: segment max ----------
__global__ void k_edge_max(const int* __restrict__ src, const int* __restrict__ dst,
                           const float* __restrict__ ea, const float* __restrict__ aS,
                           const float* __restrict__ aD, const float* __restrict__ wEff,
                           const float* __restrict__ aLoop, unsigned* __restrict__ amax,
                           int E, int N, int Hh) {
  int i = blockIdx.x * blockDim.x + threadIdx.x;
  if (i >= E + N) return;
  bool isLoop = i >= E;
  int s, d; float eav[ED_F] = {};
  if (isLoop) { s = d = i - E; }
  else {
    s = src[i]; d = dst[i];
#pragma unroll
    for (int k = 0; k < ED_F; ++k) eav[k] = ea[(size_t)i * ED_F + k];
  }
  for (int h = 0; h < Hh; ++h) {
    float al = edge_alpha(h, Hh, s, d, eav, isLoop, aS, aD, wEff, aLoop);
    atomicMax(&amax[(size_t)d * Hh + h], mapf(al));
  }
}

// ---------- pass 2: softmax denominator ----------
__global__ void k_edge_den(const int* __restrict__ src, const int* __restrict__ dst,
                           const float* __restrict__ ea, const float* __restrict__ aS,
                           const float* __restrict__ aD, const float* __restrict__ wEff,
                           const float* __restrict__ aLoop, const unsigned* __restrict__ amax,
                           float* __restrict__ denom, int E, int N, int Hh) {
  int i = blockIdx.x * blockDim.x + threadIdx.x;
  if (i >= E + N) return;
  bool isLoop = i >= E;
  int s, d; float eav[ED_F] = {};
  if (isLoop) { s = d = i - E; }
  else {
    s = src[i]; d = dst[i];
#pragma unroll
    for (int k = 0; k < ED_F; ++k) eav[k] = ea[(size_t)i * ED_F + k];
  }
  for (int h = 0; h < Hh; ++h) {
    float al = edge_alpha(h, Hh, s, d, eav, isLoop, aS, aD, wEff, aLoop);
    float p = expf(al - unmapf(amax[(size_t)d * Hh + h]));
    atomicAdd(&denom[(size_t)d * Hh + h], p);
  }
}

// ---------- pass 3: scatter-aggregate alpha * h[src] into out[dst] ----------
__global__ void k_edge_agg(const int* __restrict__ src, const int* __restrict__ dst,
                           const float* __restrict__ ea, const float* __restrict__ aS,
                           const float* __restrict__ aD, const float* __restrict__ wEff,
                           const float* __restrict__ aLoop, const unsigned* __restrict__ amax,
                           const float* __restrict__ denom, const float* __restrict__ hfeat,
                           float* __restrict__ outac, int E, int N, int Hh, int C) {
  size_t idx = (size_t)blockIdx.x * blockDim.x + threadIdx.x;
  size_t tot = (size_t)(E + N) * Hh;
  if (idx >= tot) return;
  int h = (int)(idx % Hh);
  int i = (int)(idx / Hh);
  bool isLoop = i >= E;
  int s, d; float eav[ED_F] = {};
  if (isLoop) { s = d = i - E; }
  else {
    s = src[i]; d = dst[i];
#pragma unroll
    for (int k = 0; k < ED_F; ++k) eav[k] = ea[(size_t)i * ED_F + k];
  }
  float al = edge_alpha(h, Hh, s, d, eav, isLoop, aS, aD, wEff, aLoop);
  float p = expf(al - unmapf(amax[(size_t)d * Hh + h]));
  float coef = p / (denom[(size_t)d * Hh + h] + 1e-16f);
  // gather h[src] with 16B vector loads (bases are 32-float aligned), scatter atomically
  const float4* hv = (const float4*)(hfeat + (size_t)s * (Hh * C) + h * C);
  float* ov = outac + (size_t)d * (Hh * C) + h * C;
  for (int c4 = 0; c4 < C / 4; ++c4) {
    float4 v = hv[c4];
    atomicAdd(&ov[c4 * 4 + 0], v.x * coef);
    atomicAdd(&ov[c4 * 4 + 1], v.y * coef);
    atomicAdd(&ov[c4 * 4 + 2], v.z * coef);
    atomicAdd(&ov[c4 * 4 + 3], v.w * coef);
  }
}

// ---------- BatchNorm stats (sum / sumsq per column) ----------
__global__ void k_bn_stats(const float* __restrict__ x, float* __restrict__ sums,
                           float* __restrict__ sumsq, int N, int F) {
  int c = threadIdx.x;             // blockDim.x == F
  float s = 0.f, q = 0.f;
  for (int n = blockIdx.x; n < N; n += gridDim.x) {
    float v = x[(size_t)n * F + c];
    s += v; q += v * v;
  }
  atomicAdd(&sums[c], s);
  atomicAdd(&sumsq[c], q);
}

// ---------- BN + ELU, emitting f16 activations (next layer's GEMM input) ----------
__global__ void k_bn_elu_f16(const float* __restrict__ x, const float* __restrict__ sums,
                             const float* __restrict__ sumsq, const float* __restrict__ g,
                             const float* __restrict__ b, _Float16* __restrict__ yh,
                             int N, int F) {
  size_t idx = (size_t)blockIdx.x * blockDim.x + threadIdx.x;
  if (idx >= (size_t)N * F) return;
  int c = (int)(idx % F);
  float mu = sums[c] / (float)N;
  float var = sumsq[c] / (float)N - mu * mu;
  float v = (x[idx] - mu) * rsqrtf(var + 1e-5f) * g[c] + b[c];
  v = v > 0.f ? v : (expf(v) - 1.0f);
  yh[idx] = (_Float16)v;
}

__global__ void k_elu_only(const float* __restrict__ x, float* __restrict__ y, size_t n) {
  size_t idx = (size_t)blockIdx.x * blockDim.x + threadIdx.x;
  if (idx >= n) return;
  float v = x[idx];
  y[idx] = v > 0.f ? v : (expf(v) - 1.0f);
}

// ---------- masked segment-max pooling ----------
__global__ void k_pool(const float* __restrict__ h3, const int* __restrict__ batch,
                       const int* __restrict__ mask, unsigned* __restrict__ poolU,
                       int N, int F) {
  size_t idx = (size_t)blockIdx.x * blockDim.x + threadIdx.x;
  if (idx >= (size_t)N * F) return;
  int n = (int)(idx / F), c = (int)(idx % F);
  if (mask[n] != 1) return;
  atomicMax(&poolU[(size_t)batch[n] * F + c], mapf(h3[idx]));
}

__global__ void k_pool_fin(const unsigned* __restrict__ poolU, float* __restrict__ out, int n) {
  int idx = blockIdx.x * blockDim.x + threadIdx.x;
  if (idx >= n) return;
  unsigned u = poolU[idx];
  float v = (u == 0u) ? 0.0f : unmapf(u);
  if (!isfinite(v)) v = 0.0f;
  out[idx] = v;
}

extern "C" void kernel_launch(void* const* d_in, const int* in_sizes, int n_in,
                              void* d_out, int out_size, void* d_ws, size_t ws_size,
                              hipStream_t stream) {
  (void)n_in; (void)ws_size;
  const float* x   = (const float*)d_in[0];
  const int*   ei  = (const int*)d_in[1];
  const float* ea  = (const float*)d_in[2];
  const int*   bat = (const int*)d_in[3];
  const int*   msk = (const int*)d_in[4];
  const float* W1  = (const float*)d_in[6];
  const float* as1 = (const float*)d_in[7];
  const float* ad1 = (const float*)d_in[8];
  const float* We1 = (const float*)d_in[9];
  const float* ae1 = (const float*)d_in[10];
  const float* b1  = (const float*)d_in[11];
  const float* g1  = (const float*)d_in[12];
  const float* be1 = (const float*)d_in[13];
  const float* W2  = (const float*)d_in[14];
  const float* as2 = (const float*)d_in[15];
  const float* ad2 = (const float*)d_in[16];
  const float* We2 = (const float*)d_in[17];
  const float* ae2 = (const float*)d_in[18];
  const float* b2  = (const float*)d_in[19];
  const float* g2  = (const float*)d_in[20];
  const float* be2 = (const float*)d_in[21];
  const float* W3  = (const float*)d_in[22];
  const float* as3 = (const float*)d_in[23];
  const float* ad3 = (const float*)d_in[24];
  const float* We3 = (const float*)d_in[25];
  const float* ae3 = (const float*)d_in[26];
  const float* b3  = (const float*)d_in[27];

  const int N    = in_sizes[0] / IN_F;       // 50000
  const int E    = in_sizes[1] / 2;          // 1000000
  const int B    = out_size / OUT_F;         // 64
  const int Npad = ((N + 63) / 64) * 64;     // 64-row strips, guard-free GEMM
  const int* srcI = ei;
  const int* dstI = ei + E;

  // ---- workspace layout (float units) ----
  float* ws = (float*)d_ws;
  size_t o = 0;
  float*    hfeat = ws + o; o += (size_t)Npad * HID_F;      // padded GEMM output
  float*    outac = ws + o; o += (size_t)N * HID_F;         // aggregation accumulator
  _Float16* Ah    = (_Float16*)(ws + o); o += (size_t)Npad * (HID_F / 2); // f16 activations
  _Float16* Wt    = (_Float16*)(ws + o); o += (HID_F * HID_F) / 2;        // f16 W^T (padded)
  float*    aS    = ws + o; o += (size_t)N * HEADS;
  float*    aD    = ws + o; o += (size_t)N * HEADS;
  unsigned* amax  = (unsigned*)(ws + o); o += (size_t)N * HEADS;
  float*    denom = ws + o; o += (size_t)N * HEADS;
  float*    bnS   = ws + o; o += HID_F;       // bnS/bnQ contiguous -> one memset
  float*    bnQ   = ws + o; o += HID_F;
  float*    eaSum = ws + o; o += 8;
  float*    wEff  = ws + o; o += 32;
  float*    aLoop = ws + o; o += 8;
  float*    act3  = ws + o; o += (size_t)N * OUT_F;         // elu(h3) for pooling
  unsigned* poolU = (unsigned*)(ws + o); o += (size_t)B * OUT_F;

  // zero the f16 activation buffer once: guarantees zero K-padding (layer1 view
  // [Npad,32]) and zero M-pad rows (layer2/3 view [Npad,128], rows >= N untouched)
  hipMemsetAsync(Ah, 0, (size_t)Npad * HID_F * sizeof(_Float16), stream);
  hipMemsetAsync(eaSum, 0, 8 * sizeof(float), stream);
  k_ea_sum<<<256, 256, 0, stream>>>(ea, eaSum, E);

  auto post_gemm = [&](int P, int Hh, int C, const float* attS, const float* attD,
                       const float* We, const float* attE, const float* bias) {
    k_node_attn<<<(N * Hh + 255) / 256, 256, 0, stream>>>(hfeat, attS, attD, aS, aD, N, Hh, C);
    k_weff<<<1, 64, 0, stream>>>(We, attE, eaSum, wEff, aLoop, Hh, C, E);
    k_init_bias<<<(unsigned)(((size_t)N * P + 255) / 256), 256, 0, stream>>>(outac, bias, N, P);
    hipMemsetAsync(amax, 0, (size_t)N * Hh * sizeof(unsigned), stream);
    hipMemsetAsync(denom, 0, (size_t)N * Hh * sizeof(float), stream);
    int tot = E + N;
    k_edge_max<<<(tot + 255) / 256, 256, 0, stream>>>(srcI, dstI, ea, aS, aD, wEff, aLoop,
                                                      amax, E, N, Hh);
    k_edge_den<<<(tot + 255) / 256, 256, 0, stream>>>(srcI, dstI, ea, aS, aD, wEff, aLoop,
                                                      amax, denom, E, N, Hh);
    size_t tot3 = (size_t)tot * Hh;
    k_edge_agg<<<(unsigned)((tot3 + 255) / 256), 256, 0, stream>>>(
        srcI, dstI, ea, aS, aD, wEff, aLoop, amax, denom, hfeat, outac, E, N, Hh, C);
  };

  const int strips = Npad / 64;

  // ---- layer 1: 25 -> 4x32, BN + ELU ----
  k_cvt_x<<<(Npad * 32 + 255) / 256, 256, 0, stream>>>(x, Ah, N, Npad);
  k_cvt_w<<<(HID_F * 32 + 255) / 256, 256, 0, stream>>>(W1, Wt, IN_F, HID_F, 32);
  k_gemm_wmma<32, HID_F><<<dim3((strips * (HID_F / 16) + 3) / 4), dim3(128), 0, stream>>>(
      Ah, Wt, hfeat, Npad);
  post_gemm(HID_F, HEADS, HID_F / HEADS, as1, ad1, We1, ae1, b1);
  hipMemsetAsync(bnS, 0, 2 * HID_F * sizeof(float), stream);
  k_bn_stats<<<256, HID_F, 0, stream>>>(outac, bnS, bnQ, N, HID_F);
  k_bn_elu_f16<<<(unsigned)(((size_t)N * HID_F + 255) / 256), 256, 0, stream>>>(
      outac, bnS, bnQ, g1, be1, Ah, N, HID_F);

  // ---- layer 2: 128 -> 4x32, BN + ELU ----
  k_cvt_w<<<(HID_F * HID_F + 255) / 256, 256, 0, stream>>>(W2, Wt, HID_F, HID_F, HID_F);
  k_gemm_wmma<HID_F, HID_F><<<dim3((strips * (HID_F / 16) + 3) / 4), dim3(128), 0, stream>>>(
      Ah, Wt, hfeat, Npad);
  post_gemm(HID_F, HEADS, HID_F / HEADS, as2, ad2, We2, ae2, b2);
  hipMemsetAsync(bnS, 0, 2 * HID_F * sizeof(float), stream);
  k_bn_stats<<<256, HID_F, 0, stream>>>(outac, bnS, bnQ, N, HID_F);
  k_bn_elu_f16<<<(unsigned)(((size_t)N * HID_F + 255) / 256), 256, 0, stream>>>(
      outac, bnS, bnQ, g2, be2, Ah, N, HID_F);

  // ---- layer 3: 128 -> 1x64, ELU ----
  k_cvt_w<<<(OUT_F * HID_F + 255) / 256, 256, 0, stream>>>(W3, Wt, HID_F, OUT_F, HID_F);
  k_gemm_wmma<HID_F, OUT_F><<<dim3((strips * (OUT_F / 16) + 3) / 4), dim3(128), 0, stream>>>(
      Ah, Wt, hfeat, Npad);
  post_gemm(OUT_F, 1, OUT_F, as3, ad3, We3, ae3, b3);
  k_elu_only<<<(unsigned)(((size_t)N * OUT_F + 255) / 256), 256, 0, stream>>>(
      outac, act3, (size_t)N * OUT_F);

  // ---- GRIN pooling: segment max over mask==1 nodes ----
  hipMemsetAsync(poolU, 0, (size_t)B * OUT_F * sizeof(unsigned), stream);
  k_pool<<<(unsigned)(((size_t)N * OUT_F + 255) / 256), 256, 0, stream>>>(
      act3, bat, msk, poolU, N, OUT_F);
  k_pool_fin<<<(B * OUT_F + 255) / 256, 256, 0, stream>>>(poolU, (float*)d_out, B * OUT_F);
}
